// Encoder_59287728554195
// MI455X (gfx1250) — compile-verified, matrix operand order checked by reading, not run on hardware
//
#include <hip/hip_runtime.h>

// Problem constants (match reference)
#define N_NODES_C 100000
#define IN_CH_C   256
#define OUT_CH_C  128
#define N_EDGES_C 3200000
#define SCALE_C   1.8f

typedef __attribute__((ext_vector_type(16))) _Float16 v16h;
typedef __attribute__((ext_vector_type(8)))  _Float16 v8h;
typedef __attribute__((ext_vector_type(8)))  float    v8f;

// ---------------------------------------------------------------------------
// Degree: deg[c] = (#edges with col==c) + 1 (self loop). Native u32 atomics.
// ---------------------------------------------------------------------------
__global__ __launch_bounds__(256) void k_deg_init(unsigned* __restrict__ deg) {
  int i = blockIdx.x * 256 + threadIdx.x;
  if (i < N_NODES_C) deg[i] = 1u;  // self loop
}

__global__ __launch_bounds__(256) void k_deg_count(const long long* __restrict__ ei,
                                                   unsigned* __restrict__ deg) {
  int e = blockIdx.x * 256 + threadIdx.x;
  if (e < N_EDGES_C) {
    int col = (int)ei[(size_t)N_EDGES_C + e];
    atomicAdd(&deg[col], 1u);           // global_atomic_add_u32
  }
}

__global__ __launch_bounds__(256) void k_dinv(const unsigned* __restrict__ deg,
                                              float* __restrict__ dinv) {
  int i = blockIdx.x * 256 + threadIdx.x;
  if (i < N_NODES_C) dinv[i] = rsqrtf((float)deg[i]);   // deg >= 1 always
}

// ---------------------------------------------------------------------------
// One-time weight prep: W[k][c] (f32 row-major) -> Wt[c][k] (f16, c-major).
// 64 KB per matrix; lets every B-fragment load be a contiguous b128.
// ---------------------------------------------------------------------------
__global__ __launch_bounds__(256) void k_prep_w(const float* __restrict__ W1,
                                                const float* __restrict__ W2,
                                                _Float16* __restrict__ Wt1,
                                                _Float16* __restrict__ Wt2) {
  const int i = blockIdx.x * 256 + threadIdx.x;   // 32768 threads
  const int c = i >> 8;                           // 0..127
  const int k = i & 255;                          // 0..255
  Wt1[(size_t)c * IN_CH_C + k] = (_Float16)W1[(size_t)k * OUT_CH_C + c];
  Wt2[(size_t)c * IN_CH_C + k] = (_Float16)W2[(size_t)k * OUT_CH_C + c];
}

// ---------------------------------------------------------------------------
// Fused dual GEMM + bias + branch-2 row L2-normalize*1.8.
// One workgroup = 16 rows x 128 cols. 8 waves, each owns one 16x16 N-tile.
// K=256 in 8 steps of 32 via v_wmma_f32_16x16x32_f16; A fragment shared
// between the W1 and W2 accumulators.
// Fragment layouts per CDNA5 ISA 7.12.2:
//   A  (16x32 f16): lane L<16 -> row L, elems[0..7]=K(kb+0..7), [8..15]=K(kb+16..23)
//                   lane 16+L -> row L, K(kb+8..15), K(kb+24..31)
//   B  (32x16 f16): mirrored with N=lane&15 (k-major per lane)
//   C/D (16x16 f32): elem v -> row v (lanes 0-15) / row 8+v (lanes 16-31), N=lane&15
// ---------------------------------------------------------------------------
__global__ __launch_bounds__(256) void k_gemm_dual(
    const float* __restrict__ x,
    const _Float16* __restrict__ Wt1, const float* __restrict__ b1,
    const _Float16* __restrict__ Wt2, const float* __restrict__ b2,
    float* __restrict__ y1, float* __restrict__ h) {
  __shared__ float sq[16];

  const int lane = threadIdx.x & 31;
  const int wave = threadIdx.x >> 5;
  const int m0   = blockIdx.x * 16;          // 6250 blocks * 16 = 100000 rows
  const int n0   = wave * 16;                // 8 waves * 16 = 128 cols
  const int l15  = lane & 15;
  const int koff = (lane >> 4) * 8;          // 0 for lanes 0-15, 8 for 16-31
  const int row  = m0 + l15;                 // A row this lane feeds
  const int coln = n0 + l15;                 // B column this lane feeds

  if (threadIdx.x < 16) sq[threadIdx.x] = 0.0f;
  __syncthreads();

  v8f c1 = {};
  v8f c2 = {};
  const float*    __restrict__ xrow  = x   + (size_t)row  * IN_CH_C;
  const _Float16* __restrict__ w1row = Wt1 + (size_t)coln * IN_CH_C;
  const _Float16* __restrict__ w2row = Wt2 + (size_t)coln * IN_CH_C;

#pragma unroll
  for (int kb = 0; kb < IN_CH_C; kb += 32) {
    v16h a;
#pragma unroll
    for (int i = 0; i < 8; ++i) {
      a[i]     = (_Float16)xrow[kb + koff + i];        // b128 + v_cvt_pk_f16_f32
      a[8 + i] = (_Float16)xrow[kb + 16 + koff + i];
    }
    // Contiguous 16-byte fragment loads from pre-transposed f16 weights.
    const v8h bA0 = *(const v8h*)(w1row + kb + koff);
    const v8h bA1 = *(const v8h*)(w1row + kb + 16 + koff);
    const v8h bB0 = *(const v8h*)(w2row + kb + koff);
    const v8h bB1 = *(const v8h*)(w2row + kb + 16 + koff);
    const v16h bA = __builtin_shufflevector(bA0, bA1, 0,1,2,3,4,5,6,7,8,9,10,11,12,13,14,15);
    const v16h bB = __builtin_shufflevector(bB0, bB1, 0,1,2,3,4,5,6,7,8,9,10,11,12,13,14,15);

    c1 = __builtin_amdgcn_wmma_f32_16x16x32_f16(false, a, false, bA, (short)0, c1, false, false);
    c2 = __builtin_amdgcn_wmma_f32_16x16x32_f16(false, a, false, bB, (short)0, c2, false, false);
  }

  const float bias1 = b1[coln];
  const float bias2 = b2[coln];
#pragma unroll
  for (int v = 0; v < 8; ++v) { c1[v] += bias1; c2[v] += bias2; }

  // Cross-wave per-row sum of squares for branch 2 (rows m0..m0+15).
#pragma unroll
  for (int v = 0; v < 8; ++v) {
    const int r = koff + v;                  // row index 0..15 within tile
    atomicAdd(&sq[r], c2[v] * c2[v]);        // ds_add_f32
  }
  __syncthreads();

#pragma unroll
  for (int v = 0; v < 8; ++v) {
    const int r = koff + v;
    const float s = SCALE_C / fmaxf(sqrtf(sq[r]), 1e-12f);
    const size_t o = (size_t)(m0 + r) * OUT_CH_C + n0 + l15;
    y1[o] = c1[v];
    h[o]  = c2[v] * s;
  }
}

// ---------------------------------------------------------------------------
// Self-loop term doubles as output init: out[i*C+c] = src[i*C+c] * dinv[i]^2
// (poisoned d_out gets fully overwritten here before edge accumulation).
// ---------------------------------------------------------------------------
__global__ __launch_bounds__(256) void k_selfloop_init(
    const float* __restrict__ y1, const float* __restrict__ h,
    const float* __restrict__ dinv,
    float* __restrict__ out_h, float* __restrict__ out_x) {
  const size_t i = (size_t)blockIdx.x * 256 + threadIdx.x;   // N*128 threads
  const int node = (int)(i >> 7);
  const float d = dinv[node];
  const float w = d * d;
  out_h[i] = h[i]  * w;
  out_x[i] = y1[i] * w;
}

// ---------------------------------------------------------------------------
// Edge scatter: one wave32 per edge, 4 channels/lane (float4 gather).
// L2 (192MB) holds y1+h+outputs (~205MB) nearly resident -> random traffic
// stays on-chip. Hardware global_atomic_add_f32 via unsafeAtomicAdd.
// ---------------------------------------------------------------------------
__global__ __launch_bounds__(256) void k_propagate_edges(
    const long long* __restrict__ ei,
    const float* __restrict__ dinv,
    const float* __restrict__ y1, const float* __restrict__ h,
    float* __restrict__ out_x, float* __restrict__ out_h) {
  const int lane = threadIdx.x & 31;
  const long long e = (long long)blockIdx.x * 8 + (threadIdx.x >> 5);
  if (e >= N_EDGES_C) return;

  // Stream-ahead prefetch of the edge list (global_prefetch_b8).
  __builtin_prefetch(ei + e + 4096, 0, 0);

  const int row = (int)ei[e];
  const int col = (int)ei[(size_t)N_EDGES_C + e];
  const float norm = dinv[row] * dinv[col];

  const size_t src = (size_t)row * OUT_CH_C + lane * 4;
  const size_t dst = (size_t)col * OUT_CH_C + lane * 4;

  const float4 vh = *(const float4*)(h  + src);
  const float4 vx = *(const float4*)(y1 + src);

  unsafeAtomicAdd(out_h + dst + 0, vh.x * norm);
  unsafeAtomicAdd(out_h + dst + 1, vh.y * norm);
  unsafeAtomicAdd(out_h + dst + 2, vh.z * norm);
  unsafeAtomicAdd(out_h + dst + 3, vh.w * norm);

  unsafeAtomicAdd(out_x + dst + 0, vx.x * norm);
  unsafeAtomicAdd(out_x + dst + 1, vx.y * norm);
  unsafeAtomicAdd(out_x + dst + 2, vx.z * norm);
  unsafeAtomicAdd(out_x + dst + 3, vx.w * norm);
}

// ---------------------------------------------------------------------------
extern "C" void kernel_launch(void* const* d_in, const int* in_sizes, int n_in,
                              void* d_out, int out_size, void* d_ws, size_t ws_size,
                              hipStream_t stream) {
  const float*     x  = (const float*)d_in[0];
  const long long* ei = (const long long*)d_in[1];   // int64 edge_index [2, E]
  const float*     W1 = (const float*)d_in[2];
  const float*     b1 = (const float*)d_in[3];
  const float*     W2 = (const float*)d_in[4];
  const float*     b2 = (const float*)d_in[5];

  // Output tuple (h, x_) concatenated flat.
  float* out_h = (float*)d_out;
  float* out_x = (float*)d_out + (size_t)N_NODES_C * OUT_CH_C;

  // Workspace layout (bytes):
  //   deg u32 | dinv f32 | y1 f32 [N,128] | h f32 [N,128] | Wt1 f16 | Wt2 f16
  char* ws = (char*)d_ws;
  const size_t featBytes = (size_t)N_NODES_C * OUT_CH_C * 4;   // 51.2 MB
  unsigned*  deg  = (unsigned*)(ws);
  float*     dinv = (float*)(ws + (size_t)N_NODES_C * 4);
  float*     y1   = (float*)(ws + (size_t)N_NODES_C * 8);
  float*     hbuf = (float*)(ws + (size_t)N_NODES_C * 8 + featBytes);
  _Float16*  Wt1  = (_Float16*)(ws + (size_t)N_NODES_C * 8 + 2 * featBytes);
  _Float16*  Wt2  = Wt1 + (size_t)OUT_CH_C * IN_CH_C;

  // 1) degrees + dinv, and weight transpose/convert (independent, same stream)
  k_deg_init <<<(N_NODES_C + 255) / 256, 256, 0, stream>>>(deg);
  k_deg_count<<<(N_EDGES_C + 255) / 256, 256, 0, stream>>>(ei, deg);
  k_dinv     <<<(N_NODES_C + 255) / 256, 256, 0, stream>>>(deg, dinv);
  k_prep_w   <<<(OUT_CH_C * IN_CH_C) / 256, 256, 0, stream>>>(W1, W2, Wt1, Wt2);

  // 2) fused dual GEMM + bias + branch-2 normalize (WMMA)
  k_gemm_dual<<<N_NODES_C / 16, 256, 0, stream>>>(x, Wt1, b1, Wt2, b2, y1, hbuf);

  // 3) self-loop term initializes outputs
  k_selfloop_init<<<((size_t)N_NODES_C * OUT_CH_C) / 256, 256, 0, stream>>>(
      y1, hbuf, dinv, out_h, out_x);

  // 4) edge scatter-accumulate (one wave per edge, both branches fused)
  k_propagate_edges<<<N_EDGES_C / 8, 256, 0, stream>>>(
      ei, dinv, y1, hbuf, out_x, out_h);
}
